// GAttn_67147518705771
// MI455X (gfx1250) — compile-verified, matrix operand order checked by reading, not run on hardware
//
#include <hip/hip_runtime.h>

typedef __attribute__((ext_vector_type(2)))  float    v2f;
typedef __attribute__((ext_vector_type(8)))  float    v8f;
typedef __attribute__((ext_vector_type(16))) _Float16 v16h;

#define LOG2E 1.4426950408889634f
#define EPS_N 0.01f
#define NWAVES 8

// ---------------------------------------------------------------------------
// prep_geo: build WMMA operands for the Gram/arg computation.
//   Ag[jb][lane] (A operand of V_WMMA_F32_16X16X4_F32, both modalities packed):
//     lane<16 : (-2c0*gf0[j][0], -2c0*gf0[j][1], -2c1*gf1[j][0], -2c1*gf1[j][1]),  j=jb*16+lane
//     lane>=16: (-2c0*gf0[j][2],  c0*|gf0[j]|^2, -2c1*gf1[j][2],  c1*|gf1[j]|^2),  j=jb*16+lane-16
//   Bg[ib][lane] (B operand): lane<16: (gf0[i][0],gf0[i][1],gf1[i][0],gf1[i][1]);
//                             lane>=16:(gf0[i][2],1, gf1[i][2],1)
//   Csq{0,1}[i] = c*|gf[i]|^2 (C-operand splat) with c = -log2e/(2*sigma^2)
// ---------------------------------------------------------------------------
__global__ void prep_geo(const float* __restrict__ gf0, const float* __restrict__ gf1,
                         const float* __restrict__ sigmas,
                         float4* __restrict__ Ag, float4* __restrict__ Bg,
                         float* __restrict__ Csq0, float* __restrict__ Csq1, int njb) {
  int idx = blockIdx.x * blockDim.x + threadIdx.x;
  if (idx >= njb * 32) return;
  int lane = idx & 31, jb = idx >> 5;
  float s0 = sigmas[0], s1 = sigmas[1];
  float c0 = -LOG2E / (2.0f * s0 * s0);
  float c1 = -LOG2E / (2.0f * s1 * s1);
  if (lane < 16) {
    int j = jb * 16 + lane;
    float a0 = gf0[j*3+0], a1 = gf0[j*3+1], a2 = gf0[j*3+2];
    float b0 = gf1[j*3+0], b1 = gf1[j*3+1], b2 = gf1[j*3+2];
    Ag[idx] = make_float4(-2.0f*c0*a0, -2.0f*c0*a1, -2.0f*c1*b0, -2.0f*c1*b1);
    Bg[idx] = make_float4(a0, a1, b0, b1);
    Csq0[j] = c0 * (a0*a0 + a1*a1 + a2*a2);
    Csq1[j] = c1 * (b0*b0 + b1*b1 + b2*b2);
  } else {
    int j = jb * 16 + lane - 16;
    float a0 = gf0[j*3+0], a1 = gf0[j*3+1], a2 = gf0[j*3+2];
    float b0 = gf1[j*3+0], b1 = gf1[j*3+1], b2 = gf1[j*3+2];
    float sq0 = a0*a0 + a1*a1 + a2*a2;
    float sq1 = b0*b0 + b1*b1 + b2*b2;
    Ag[idx] = make_float4(-2.0f*c0*a2, c0*sq0, -2.0f*c1*b2, c1*sq1);
    Bg[idx] = make_float4(a2, 1.0f, b2, 1.0f);
  }
}

// ---------------------------------------------------------------------------
// prep_v: V (f32, Nx128) -> f16 B-operand tiles for V_WMMA_F32_16X16X32_F16,
// padded to 144 cols where col 128 == 1.0 (row-sum column), cols 129..143 == 0.
// Layout: Bh[chunk][tile(9)][lane(32)] = v16h with element e ->
//   K = (e<8 ? e : e+8) + (lane&16 ? 8 : 0), N = lane&15  (mirrors f16 A layout)
// ---------------------------------------------------------------------------
__global__ void prep_v(const float* __restrict__ V, v16h* __restrict__ Bh, int nchunks) {
  int idx = blockIdx.x * blockDim.x + threadIdx.x;
  if (idx >= nchunks * 9 * 32) return;
  int lane = idx & 31;
  int t    = (idx >> 5) % 9;
  int c    = idx / (9 * 32);
  int col  = t * 16 + (lane & 15);
  int khalf = (lane & 16) ? 8 : 0;
  v16h o;
#pragma unroll
  for (int e = 0; e < 16; ++e) {
    int K = ((e < 8) ? e : e + 8) + khalf;
    int row = c * 32 + K;
    float v = (col < 128) ? V[(size_t)row * 128 + col] : ((col == 128) ? 1.0f : 0.0f);
    o[e] = (_Float16)v;
  }
  Bh[idx] = o;
}

// ---------------------------------------------------------------------------
// Main fused kernel: 1 block (8 waves) per 16-row block; split-K x8 over the
// column chunks of 32 (4096 waves total for latency hiding). Per chunk:
//   4x v_wmma_f32_16x16x4_f32  -> exp2 args for both modalities (transposed
//       Gram tile lands directly in the f16 A-operand layout)
//   v_exp_f32 + packed weighted sum + f16 pack
//   9x v_wmma_f32_16x16x32_f16 -> accumulate S@[V|1]
// Note: d2 = ||xi-xj||^2 >= 0 exactly, so the reference's maximum(d2, 0) only
// guards rounding noise; we skip it (saves a canonicalize+min per element).
// Deterministic per-wave LDS reduction, then normalize + residual.
// ---------------------------------------------------------------------------
__global__ __launch_bounds__(NWAVES * 32) void gattn_main(
    const float4* __restrict__ Ag, const float4* __restrict__ Bg,
    const float* __restrict__ Csq0, const float* __restrict__ Csq1,
    const v16h* __restrict__ Bh, const float* __restrict__ V,
    const float* __restrict__ weights, float* __restrict__ out, int nchunks) {
  __shared__ float lacc[NWAVES][16 * 144];
  const int tid  = threadIdx.x;
  const int lane = tid & 31;
  const int wave = tid >> 5;
  const int ib   = blockIdx.x;
  const int i0   = ib * 16;

  const float4 bg = Bg[ib * 32 + lane];
  const v2f bg0 = {bg.x, bg.y};
  const v2f bg1 = {bg.z, bg.w};
  const float csq0 = Csq0[i0 + (lane & 15)];
  const float csq1 = Csq1[i0 + (lane & 15)];
  v8f c0s, c1s;
#pragma unroll
  for (int r = 0; r < 8; ++r) { c0s[r] = csq0; c1s[r] = csq1; }
  const float w0 = weights[0], w1 = weights[1];

  v8f acc[9];
#pragma unroll
  for (int t = 0; t < 9; ++t) {
    v8f z = {0.f, 0.f, 0.f, 0.f, 0.f, 0.f, 0.f, 0.f};
    acc[t] = z;
  }

  // software-pipelined A operands (one chunk ahead)
  int c = wave;
  float4 ag0 = Ag[(size_t)(2 * c + 0) * 32 + lane];
  float4 ag1 = Ag[(size_t)(2 * c + 1) * 32 + lane];

  while (c < nchunks) {
    const int cn = c + NWAVES;
    const int cp = (cn < nchunks) ? cn : c;   // clamped prefetch index
    const float4 nag0 = Ag[(size_t)(2 * cp + 0) * 32 + lane];
    const float4 nag1 = Ag[(size_t)(2 * cp + 1) * 32 + lane];

    // issue all 9 B-tile loads up front; exp block below covers their latency
    const v16h* bp = Bh + (size_t)c * 9 * 32 + lane;
    v16h bt[9];
#pragma unroll
    for (int t = 0; t < 9; ++t) bt[t] = bp[t * 32];

    v2f a00 = {ag0.x, ag0.y}, a01 = {ag0.z, ag0.w};
    v2f a10 = {ag1.x, ag1.y}, a11 = {ag1.z, ag1.w};
    // D(M=j, N=i) = -2c*dot(gf_j, gf_i) + c*|gf_j|^2 + c*|gf_i|^2  (log2-domain arg)
    v8f d00 = __builtin_amdgcn_wmma_f32_16x16x4_f32(false, a00, false, bg0, (short)0, c0s, false, false);
    v8f d01 = __builtin_amdgcn_wmma_f32_16x16x4_f32(false, a01, false, bg1, (short)0, c1s, false, false);
    v8f d10 = __builtin_amdgcn_wmma_f32_16x16x4_f32(false, a10, false, bg0, (short)0, c0s, false, false);
    v8f d11 = __builtin_amdgcn_wmma_f32_16x16x4_f32(false, a11, false, bg1, (short)0, c1s, false, false);

    v16h Ah;
#pragma unroll
    for (int r = 0; r < 8; ++r) {
      float e0 = __builtin_amdgcn_exp2f(d00[r]);
      float e1 = __builtin_amdgcn_exp2f(d01[r]);
      Ah[r] = (_Float16)(w0 * e0 + w1 * e1);
      float f0 = __builtin_amdgcn_exp2f(d10[r]);
      float f1 = __builtin_amdgcn_exp2f(d11[r]);
      Ah[8 + r] = (_Float16)(w0 * f0 + w1 * f1);
    }

#pragma unroll
    for (int t = 0; t < 9; ++t) {
      acc[t] = __builtin_amdgcn_wmma_f32_16x16x32_f16(false, Ah, false, bt[t],
                                                      (short)0, acc[t], false, false);
    }

    ag0 = nag0;
    ag1 = nag1;
    c = cn;
  }

  // each wave writes its full 16x144 partial (every slot written exactly once)
#pragma unroll
  for (int t = 0; t < 9; ++t) {
#pragma unroll
    for (int r = 0; r < 8; ++r) {
      int i   = r + 8 * (lane >> 4);
      int col = t * 16 + (lane & 15);
      lacc[wave][i * 144 + col] = acc[t][r];
    }
  }
  __syncthreads();

  // normalize + residual; deterministic NWAVES-way sum
  for (int k = tid; k < 16 * 128; k += NWAVES * 32) {
    int row = k >> 7, col = k & 127;
    float o = 0.0f, s = 0.0f;
#pragma unroll
    for (int w = 0; w < NWAVES; ++w) {
      o += lacc[w][row * 144 + col];
      s += lacc[w][row * 144 + 128];
    }
    size_t oi = (size_t)(i0 + row) * 128 + col;
    out[oi] = o / (s + EPS_N) + V[oi];
  }
}

extern "C" void kernel_launch(void* const* d_in, const int* in_sizes, int n_in,
                              void* d_out, int out_size, void* d_ws, size_t ws_size,
                              hipStream_t stream) {
  const float* gf0     = (const float*)d_in[0];
  const float* gf1     = (const float*)d_in[1];
  const float* V       = (const float*)d_in[2];
  const float* weights = (const float*)d_in[3];
  const float* sigmas  = (const float*)d_in[4];
  float* out = (float*)d_out;

  const int N       = in_sizes[2] / 128;  // 8192
  const int njb     = N / 16;             // 512 16-row/col blocks
  const int nchunks = N / 32;             // 256 column chunks

  char* ws = (char*)d_ws;
  size_t szAg = (size_t)njb * 32 * sizeof(float4);           // 256 KB
  float4* Ag  = (float4*)(ws);
  float4* Bg  = (float4*)(ws + szAg);
  float*  Cs0 = (float*)(ws + 2 * szAg);
  float*  Cs1 = Cs0 + N;
  v16h*   Bh  = (v16h*)(ws + 2 * szAg + 2 * (size_t)N * sizeof(float));  // ~2.25 MB

  prep_geo<<<(njb * 32 + 255) / 256, 256, 0, stream>>>(gf0, gf1, sigmas, Ag, Bg, Cs0, Cs1, njb);
  prep_v<<<(nchunks * 9 * 32 + 255) / 256, 256, 0, stream>>>(V, Bh, nchunks);
  gattn_main<<<njb, NWAVES * 32, 0, stream>>>(Ag, Bg, Cs0, Cs1, Bh, V, weights, out, nchunks);
}
